// MHSA_72636486909982
// MI455X (gfx1250) — matrix operand order repaired and checked by reading, not compiled, not run on hardware
//
#include <hip/hip_runtime.h>
#include <hip/hip_bf16.h>

// MHSA for MI455X (gfx1250): f16 WMMA with f32 accumulation.
// Kernel 0: f32 -> f16 convert (x, Wq, Wk, Wv) so GEMMs stream f16.
// Kernel 1: QKV projection GEMM, 128x128 tiles, 8 waves, double-buffered LDS
//           fed by GLOBAL_LOAD_ASYNC_TO_LDS_B128; V stored transposed.
// Kernel 2: flash attention, 128 query rows/block, async double-buffered K/V^T.

typedef _Float16 f16;
typedef __attribute__((ext_vector_type(4)))  _Float16 v4h;
typedef __attribute__((ext_vector_type(8)))  _Float16 v8h;
typedef __attribute__((ext_vector_type(16))) _Float16 v16h;
typedef __attribute__((ext_vector_type(8)))  float    v8f;
typedef int i32x4 __attribute__((vector_size(16)));   // matches builtin param type

constexpr int kDIM   = 1024;
constexpr int kSEQ   = 2048;
constexpr int kBATCH = 2;
constexpr int kHD    = 64;
constexpr int kBSM   = kBATCH * kSEQ;   // 4096 rows

#if __has_builtin(__builtin_amdgcn_global_load_async_to_lds_b128) && \
    __has_builtin(__builtin_amdgcn_s_wait_asynccnt)
#define HAVE_ASYNC_LDS 1
#else
#define HAVE_ASYNC_LDS 0
#endif

static __device__ __forceinline__ void copy16_to_lds(f16* lds_dst, const f16* gsrc) {
#if HAVE_ASYNC_LDS
  __builtin_amdgcn_global_load_async_to_lds_b128(
      (__attribute__((address_space(1))) i32x4*)gsrc,
      (__attribute__((address_space(3))) i32x4*)lds_dst, 0, 0);
#else
  *(v8h*)lds_dst = *(const v8h*)gsrc;
#endif
}

static __device__ __forceinline__ void wait_async_copies() {
#if HAVE_ASYNC_LDS
  __builtin_amdgcn_s_wait_asynccnt(0);
#endif
}

static __device__ __forceinline__ v8f wmma_f16(v16h a, v16h b, v8f c) {
  return __builtin_amdgcn_wmma_f32_16x16x32_f16(false, a, false, b, (short)0, c,
                                                false, false);
}

static __device__ __forceinline__ v16h pack16(v8h lo, v8h hi8) {
  v16h r;
#pragma unroll
  for (int i = 0; i < 8; ++i) { r[i] = lo[i]; r[8 + i] = hi8[i]; }
  return r;
}

// ---------------------------------------------------------------------------
// Kernel 0: f32 -> f16, 8 elements per thread.
// ---------------------------------------------------------------------------
__global__ __launch_bounds__(256)
void mhsa_cvt_f16(const float* __restrict__ src, f16* __restrict__ dst, int n) {
  int i = (blockIdx.x * 256 + threadIdx.x) * 8;
  if (i >= n) return;
  float4 a = *(const float4*)(src + i);
  float4 b = *(const float4*)(src + i + 4);
  v8h h = { (f16)a.x, (f16)a.y, (f16)a.z, (f16)a.w,
            (f16)b.x, (f16)b.y, (f16)b.z, (f16)b.w };
  *(v8h*)(dst + i) = h;
}

// ---------------------------------------------------------------------------
// Kernel 1: y[m,n] = sum_k x[m,k] W[n,k] + b[n].
// Block 256 threads (8 waves); tile M=128, N=128, BK=32.
// Wave (wm = w&3, wn = w>>2) owns a 32x64 strip -> acc[2][4].
// z==2 (V) is stored transposed: vth[n][m], contiguous v8h per fragment.
// ---------------------------------------------------------------------------
__global__ __launch_bounds__(256)
void mhsa_qkv_gemm(const f16* __restrict__ xh,
                   const f16* __restrict__ wqh, const f16* __restrict__ wkh,
                   const f16* __restrict__ wvh,
                   const float* __restrict__ bq, const float* __restrict__ bk,
                   const float* __restrict__ bv,
                   f16* __restrict__ qh, f16* __restrict__ kh,
                   f16* __restrict__ vth)
{
  __shared__ __align__(16) f16 lx[2][128][40];  // 10KB per buffer
  __shared__ __align__(16) f16 lw[2][128][40];

  const int tid  = threadIdx.x;
  const int wave = tid >> 5;
  const int lane = tid & 31;
  const int lrow = lane & 15;
  const int lhi  = lane >> 4;
  const int wm = wave & 3;      // M subtile (4 x 32 rows)
  const int wn = wave >> 2;     // N subtile (2 x 64 cols)
  const int m0 = blockIdx.x * 128;
  const int n0 = blockIdx.y * 128;
  const int z  = blockIdx.z;

  const f16*   W  = (z == 0) ? wqh : (z == 1) ? wkh : wvh;
  const float* bs = (z == 0) ? bq  : (z == 1) ? bk  : bv;

  const int khalf = lhi * 8;    // A-fragment k sub-offset
  const int kb    = lhi * 16;   // unused; B fragments also use khalf

  v8f acc[2][4] = {};

  auto issue = [&](int buf, int k0) {
#pragma unroll
    for (int t = 0; t < 2; ++t) {
      int idx = tid + t * 256;          // 512 16B chunks per tile
      int r = idx >> 2, c = idx & 3;    // 4 chunks per 32-half row
      copy16_to_lds(&lx[buf][r][c * 8], xh + (size_t)(m0 + r) * kDIM + k0 + c * 8);
      copy16_to_lds(&lw[buf][r][c * 8], W  + (size_t)(n0 + r) * kDIM + k0 + c * 8);
    }
  };

  issue(0, 0);
  for (int ks = 0; ks < kDIM / 32; ++ks) {
    wait_async_copies();
    __syncthreads();
    if (ks + 1 < kDIM / 32) issue((ks + 1) & 1, (ks + 1) * 32);

    const int buf = ks & 1;
    const f16* ar0 = &lx[buf][32 * wm + lrow][0];
    const f16* ar1 = &lx[buf][32 * wm + 16 + lrow][0];
    v16h a0 = pack16(*(const v8h*)(ar0 + khalf), *(const v8h*)(ar0 + 16 + khalf));
    v16h a1 = pack16(*(const v8h*)(ar1 + khalf), *(const v8h*)(ar1 + 16 + khalf));

#pragma unroll
    for (int j = 0; j < 4; ++j) {
      const f16* brow = &lw[buf][64 * wn + 16 * j + lrow][0];
      v16h bf = pack16(*(const v8h*)(brow + khalf), *(const v8h*)(brow + 16 + khalf));
      acc[0][j] = wmma_f16(a0, bf, acc[0][j]);
      acc[1][j] = wmma_f16(a1, bf, acc[1][j]);
    }
  }

#pragma unroll
  for (int mi = 0; mi < 2; ++mi) {
#pragma unroll
    for (int j = 0; j < 4; ++j) {
      int n = n0 + 64 * wn + 16 * j + lrow;
      float bb = bs[n];
      if (z == 2) {
        // V^T store: C-fragment rows r+8*lhi are consecutive m -> one v8h.
        v8h hv;
#pragma unroll
        for (int r = 0; r < 8; ++r) hv[r] = (f16)(acc[mi][j][r] + bb);
        int m = m0 + 32 * wm + 16 * mi + 8 * lhi;
        *(v8h*)(vth + (size_t)n * kBSM + m) = hv;
      } else {
        f16* outp = (z == 0) ? qh : kh;
#pragma unroll
        for (int r = 0; r < 8; ++r) {
          int m = m0 + 32 * wm + 16 * mi + r + 8 * lhi;
          outp[(size_t)m * kDIM + n] = (f16)(acc[mi][j][r] + bb);
        }
      }
    }
  }
}

// ---------------------------------------------------------------------------
// Kernel 2: flash attention per (batch, head).
// Block 256 threads (8 waves) = 128 query rows; 64-row key tiles,
// double-buffered async K / V^T loads, online softmax.
// ---------------------------------------------------------------------------
__global__ __launch_bounds__(256)
void mhsa_attention(const f16* __restrict__ qh, const f16* __restrict__ kh,
                    const f16* __restrict__ vth, float* __restrict__ out)
{
  __shared__ __align__(16) f16 kt[2][64][72];    // K tile: [nk][d]
  __shared__ __align__(16) f16 vt[2][64][72];    // V^T tile: [d][nk]
  __shared__ __align__(16) f16 pt[8][16][72];    // per-wave P bounce

  const int tid  = threadIdx.x;
  const int wave = tid >> 5;
  const int lane = tid & 31;
  const int lrow = lane & 15;
  const int lhi  = lane >> 4;
  const int q0 = blockIdx.x * 128;
  const int bh = blockIdx.y;
  const int b = bh >> 4, h = bh & 15;
  const size_t headoff = (size_t)h * kHD;
  const size_t rowbase = (size_t)b * kSEQ;

  const int khalf = lhi * 8;
  const int kb    = lhi * 16;
  const float scale = 0.125f;   // 1/sqrt(64)

  auto issue = [&](int buf, int kt0) {
#pragma unroll
    for (int t = 0; t < 2; ++t) {
      int idx = tid + t * 256;          // 512 16B chunks per tile
      int r = idx >> 3, c = idx & 7;    // 8 chunks per 64-half row
      copy16_to_lds(&kt[buf][r][c * 8],
                    kh + (rowbase + kt0 + r) * kDIM + headoff + c * 8);
      copy16_to_lds(&vt[buf][r][c * 8],
                    vth + (headoff + r) * kBSM + rowbase + kt0 + c * 8);
    }
  };

  // Q fragments for this wave's 16 rows, held in registers.
  const f16* qrow = qh + (rowbase + q0 + 16 * wave + lrow) * kDIM + headoff;
  v16h aq0 = pack16(*(const v8h*)(qrow + khalf),      *(const v8h*)(qrow + 16 + khalf));
  v16h aq1 = pack16(*(const v8h*)(qrow + 32 + khalf), *(const v8h*)(qrow + 48 + khalf));

  v8f o[4] = {};
  float m_i[8], l_i[8];
#pragma unroll
  for (int r = 0; r < 8; ++r) { m_i[r] = -1e30f; l_i[r] = 0.f; }

  issue(0, 0);
  for (int it = 0; it < kSEQ / 64; ++it) {
    wait_async_copies();
    __syncthreads();
    if (it + 1 < kSEQ / 64) issue((it + 1) & 1, (it + 1) * 64);
    const int buf = it & 1;

    // S = scale * Q K^T (per-wave 16x64).
    v8f s[4] = {};
#pragma unroll
    for (int j = 0; j < 4; ++j) {
      const f16* brow = &kt[buf][16 * j + lrow][0];
      v16h b0 = pack16(*(const v8h*)(brow + khalf),      *(const v8h*)(brow + 16 + khalf));
      v16h b1 = pack16(*(const v8h*)(brow + 32 + khalf), *(const v8h*)(brow + 48 + khalf));
      s[j] = wmma_f16(aq0, b0, s[j]);
      s[j] = wmma_f16(aq1, b1, s[j]);
    }
#pragma unroll
    for (int j = 0; j < 4; ++j) s[j] = s[j] * scale;

    // Online softmax. Row M=r+8*lhi lives at VGPR r across a 16-lane group.
    float mnew[8], alpha[8];
#pragma unroll
    for (int r = 0; r < 8; ++r) {
      float v = fmaxf(fmaxf(s[0][r], s[1][r]), fmaxf(s[2][r], s[3][r]));
#pragma unroll
      for (int off = 8; off >= 1; off >>= 1) v = fmaxf(v, __shfl_xor(v, off, 16));
      mnew[r] = fmaxf(m_i[r], v);
      alpha[r] = __expf(m_i[r] - mnew[r]);
    }
#pragma unroll
    for (int j = 0; j < 4; ++j)
#pragma unroll
      for (int r = 0; r < 8; ++r)
        s[j][r] = __expf(s[j][r] - mnew[r]);
#pragma unroll
    for (int r = 0; r < 8; ++r) {
      float sv = s[0][r] + s[1][r] + s[2][r] + s[3][r];
#pragma unroll
      for (int off = 8; off >= 1; off >>= 1) sv += __shfl_xor(sv, off, 16);
      l_i[r] = l_i[r] * alpha[r] + sv;
      m_i[r] = mnew[r];
    }
#pragma unroll
    for (int j = 0; j < 4; ++j)
#pragma unroll
      for (int r = 0; r < 8; ++r)
        o[j][r] *= alpha[r];

    // C-layout P -> per-wave LDS -> A-layout fragments.
    // Same-wave DS ops are in-order; compiler inserts s_wait_dscnt (may-alias).
#pragma unroll
    for (int j = 0; j < 4; ++j)
#pragma unroll
      for (int r = 0; r < 8; ++r)
        pt[wave][r + 8 * lhi][16 * j + lrow] = (f16)s[j][r];

    const f16* prow = &pt[wave][lrow][0];
    v16h pa0 = pack16(*(const v8h*)(prow + khalf),      *(const v8h*)(prow + 16 + khalf));
    v16h pa1 = pack16(*(const v8h*)(prow + 32 + khalf), *(const v8h*)(prow + 48 + khalf));

    // O += P V (contraction over nk, V^T tile is contiguous in nk).
#pragma unroll
    for (int j = 0; j < 4; ++j) {
      const f16* vrow = &vt[buf][16 * j + lrow][0];
      v16h b0 = pack16(*(const v8h*)(vrow + khalf),      *(const v8h*)(vrow + 16 + khalf));
      v16h b1 = pack16(*(const v8h*)(vrow + 32 + khalf), *(const v8h*)(vrow + 48 + khalf));
      o[j] = wmma_f16(pa0, b0, o[j]);
      o[j] = wmma_f16(pa1, b1, o[j]);
    }
  }

  float invl[8];
#pragma unroll
  for (int r = 0; r < 8; ++r) invl[r] = 1.0f / l_i[r];
#pragma unroll
  for (int j = 0; j < 4; ++j) {
    int d = 16 * j + lrow;
#pragma unroll
    for (int r = 0; r < 8; ++r) {
      int srow = q0 + 16 * wave + r + 8 * lhi;
      out[(rowbase + srow) * kDIM + headoff + d] = o[j][r] * invl[r];
    }
  }
}

// ---------------------------------------------------------------------------
extern "C" void kernel_launch(void* const* d_in, const int* in_sizes, int n_in,
                              void* d_out, int out_size, void* d_ws, size_t ws_size,
                              hipStream_t stream) {
  (void)in_sizes; (void)n_in; (void)out_size; (void)ws_size;
  const float* x  = (const float*)d_in[0];
  const float* Wq = (const float*)d_in[1];
  const float* bq = (const float*)d_in[2];
  const float* Wk = (const float*)d_in[3];
  const float* bk = (const float*)d_in[4];
  const float* Wv = (const float*)d_in[5];
  const float* bv = (const float*)d_in[6];
  float* out = (float*)d_out;

  const size_t xel = (size_t)kBSM * kDIM;    // 4,194,304
  const size_t wel = (size_t)kDIM * kDIM;    // 1,048,576
  f16* xh  = (f16*)d_ws;                      // total ws use: ~40 MB of f16
  f16* wqh = xh  + xel;
  f16* wkh = wqh + wel;
  f16* wvh = wkh + wel;
  f16* qh  = wvh + wel;
  f16* kh  = qh  + xel;
  f16* vth = kh  + xel;                       // transposed: [n][m]

  mhsa_cvt_f16<<<dim3(xel / 2048), 256, 0, stream>>>(x,  xh,  (int)xel);
  mhsa_cvt_f16<<<dim3(wel / 2048), 256, 0, stream>>>(Wq, wqh, (int)wel);
  mhsa_cvt_f16<<<dim3(wel / 2048), 256, 0, stream>>>(Wk, wkh, (int)wel);
  mhsa_cvt_f16<<<dim3(wel / 2048), 256, 0, stream>>>(Wv, wvh, (int)wel);

  dim3 g1(kBSM / 128, kDIM / 128, 3);
  mhsa_qkv_gemm<<<g1, 256, 0, stream>>>(xh, wqh, wkh, wvh, bq, bk, bv,
                                        qh, kh, vth);

  dim3 g2(kSEQ / 128, kBATCH * 16);
  mhsa_attention<<<g2, 256, 0, stream>>>(qh, kh, vth, out);
}
